// ClosestEmbeddingsLayer_85641647882722
// MI455X (gfx1250) — compile-verified, matrix operand order checked by reading, not run on hardware
//
#include <hip/hip_runtime.h>
#include <hip/hip_bf16.h>

#define BATCH    4096
#define DIM      128
#define VOCAB    100000
#define VPAD     100096        // 782 * 128, catalog padded with zero rows
#define NHIST    50
#define TOPK     500
#define NBINS    128
#define ROWS     64            // batch rows per GEMM block (4 x 16-row A sets)
#define NSETS    4
#define VCHUNK   (98 * 128)    // vocab columns per GEMM block (8 chunks cover VPAD)
#define VSPLIT   8
#define CAND_CAP 1024
#define SORT_N   1024

typedef __attribute__((ext_vector_type(16))) _Float16 v16h;
typedef __attribute__((ext_vector_type(8)))  _Float16 v8h;
typedef __attribute__((ext_vector_type(4)))  _Float16 v4h;
typedef __attribute__((ext_vector_type(8)))  float    v8f;

union Frag { v16h f; v8h h[2]; };

__device__ __forceinline__ int score_bin(float s) {
    int bin = (int)((s * 0.5f + 0.5f) * (float)NBINS);
    return bin < 0 ? 0 : (bin > NBINS - 1 ? NBINS - 1 : bin);
}

// ---------------------------------------------------------------------------
// L2-normalize rows (fp32 -> f16). Rows >= nvalid become zero padding.
// One wave per row; lane handles 4 contiguous dims.
// ---------------------------------------------------------------------------
__global__ __launch_bounds__(256) void normalize_f16(const float* __restrict__ src,
                                                     _Float16* __restrict__ dst,
                                                     int nrows, int nvalid) {
    const int wave = threadIdx.x >> 5;
    const int lane = threadIdx.x & 31;
    const int row  = blockIdx.x * 8 + wave;
    if (row >= nrows) return;  // whole wave exits together

    float x0 = 0.f, x1 = 0.f, x2 = 0.f, x3 = 0.f;
    if (row < nvalid) {
        const float* sp = src + (size_t)row * DIM + lane * 4;
        x0 = sp[0]; x1 = sp[1]; x2 = sp[2]; x3 = sp[3];
    }
    float ss = x0 * x0 + x1 * x1 + x2 * x2 + x3 * x3;
#pragma unroll
    for (int m = 16; m >= 1; m >>= 1) ss += __shfl_xor(ss, m, 32);
    const float scale = (row < nvalid) ? (1.0f / fmaxf(sqrtf(ss), 1e-8f)) : 0.0f;

    v4h outv;
    outv[0] = (_Float16)(x0 * scale);
    outv[1] = (_Float16)(x1 * scale);
    outv[2] = (_Float16)(x2 * scale);
    outv[3] = (_Float16)(x3 * scale);
    *(v4h*)(dst + (size_t)row * DIM + lane * 4) = outv;
}

__global__ void zero_ints(int* __restrict__ p, int n) {
    const int i = blockIdx.x * 256 + threadIdx.x;
    if (i < n) p[i] = 0;
}

// ---------------------------------------------------------------------------
// WMMA fragment helpers (16x16x32 f16 layout):
//   A/B: lanes 0-15 hold row/col = lane, K-chunks {0..7,16..23};
//        lanes 16-31 hold row/col = lane-16, K-chunks {8..15,24..31}.
//   C/D: VGPR r, lanes 0-15 -> (M=r, N=lane); lanes 16-31 -> (M=r+8, N=lane-16).
// ---------------------------------------------------------------------------
__device__ __forceinline__ void load_frags(Frag fr[NSETS > 1 ? 4 : 4],
                                           const _Float16* __restrict__ rowp, int laneHi) {
#pragma unroll
    for (int kk = 0; kk < 4; ++kk) {
        const _Float16* p = rowp + kk * 32 + laneHi * 8;
        fr[kk].h[0] = *(const v8h*)(p);
        fr[kk].h[1] = *(const v8h*)(p + 16);
    }
}

// ---------------------------------------------------------------------------
// Pass 1: WMMA GEMM over a V-chunk, build per-row score histograms.
// Grid: (BATCH/ROWS, VSPLIT). Block: 256 threads = 8 waves.
// Each wave: 16 vocab columns vs 64 batch rows (4 WMMA A-sets, B reused 4x).
// ---------------------------------------------------------------------------
__global__ __launch_bounds__(256) void gemm_hist(const _Float16* __restrict__ gN,
                                                 const _Float16* __restrict__ cN,
                                                 int* __restrict__ ghist) {
    __shared__ int hist[ROWS * NBINS];  // 32 KB

    const int tid    = threadIdx.x;
    const int wave   = tid >> 5;
    const int lane   = tid & 31;
    const int laneLo = lane & 15;
    const int laneHi = lane >> 4;
    const int m0     = blockIdx.x * ROWS;
    const int vstart = blockIdx.y * VCHUNK;
    const int vend   = (vstart + VCHUNK) < VPAD ? (vstart + VCHUNK) : VPAD;

    for (int i = tid; i < ROWS * NBINS; i += 256) hist[i] = 0;
    __syncthreads();

    Frag a[NSETS][4];  // 64 rows x 128 K, resident in 128 VGPRs
#pragma unroll
    for (int s = 0; s < NSETS; ++s)
        load_frags(a[s], gN + (size_t)(m0 + s * 16 + laneLo) * DIM, laneHi);

    for (int vbase = vstart; vbase < vend; vbase += 128) {
        const int n = vbase + wave * 16 + laneLo;  // this lane's B row == its C column
        Frag b[4];
        load_frags(b, cN + (size_t)n * DIM, laneHi);
#pragma unroll
        for (int s = 0; s < NSETS; ++s) {
            v8f acc = {};
#pragma unroll
            for (int kk = 0; kk < 4; ++kk)
                acc = __builtin_amdgcn_wmma_f32_16x16x32_f16(
                        false, a[s][kk].f, false, b[kk].f, (short)0, acc, false, false);
            if (n < VOCAB) {
#pragma unroll
                for (int r = 0; r < 8; ++r) {
                    const int m = s * 16 + r + 8 * laneHi;
                    atomicAdd(&hist[m * NBINS + score_bin(acc[r])], 1);
                }
            }
        }
    }
    __syncthreads();

    int* grow = ghist + (size_t)m0 * NBINS;
    for (int i = tid; i < ROWS * NBINS; i += 256) {
        const int v = hist[i];
        if (v) atomicAdd(&grow[i], v);  // integer atomics: deterministic
    }
}

// ---------------------------------------------------------------------------
// Seed correction + per-row threshold. One thread per batch row.
// ---------------------------------------------------------------------------
__global__ __launch_bounds__(256) void threshold_k(const _Float16* __restrict__ gN,
                                                   const _Float16* __restrict__ cN,
                                                   const int* __restrict__ seeds,
                                                   int* __restrict__ ghist,
                                                   int* __restrict__ tbin) {
    const int row = blockIdx.x * 256 + threadIdx.x;
    if (row >= BATCH) return;
    int* hrow = ghist + (size_t)row * NBINS;
    const _Float16* gp = gN + (size_t)row * DIM;
    for (int h = 0; h < NHIST; ++h) {
        const int sv = seeds[(size_t)row * NHIST + h];
        if (sv >= 0 && sv < VOCAB) {
            const _Float16* cp = cN + (size_t)sv * DIM;
            float s = 0.f;
#pragma unroll 8
            for (int d = 0; d < DIM; ++d) s += (float)gp[d] * (float)cp[d];
            hrow[score_bin(s)] -= 1;  // row-private: no atomics needed
        }
    }
    int cum = 0, t = 0;
    for (int b = NBINS - 1; b >= 0; --b) {
        cum += hrow[b];
        if (cum >= TOPK) { t = b; break; }
    }
    tbin[row] = t;
}

// ---------------------------------------------------------------------------
// Pass 2: recompute WMMA GEMM, collect candidates >= threshold bin, drop seeds.
// Same grid as gemm_hist.
// ---------------------------------------------------------------------------
__global__ __launch_bounds__(256) void collect(const _Float16* __restrict__ gN,
                                               const _Float16* __restrict__ cN,
                                               const int* __restrict__ seeds,
                                               const int* __restrict__ tbin,
                                               int* __restrict__ cnt,
                                               int* __restrict__ candIdx,
                                               float* __restrict__ candScore) {
    __shared__ int tb[ROWS];
    __shared__ int seedL[ROWS][NHIST];  // 12.8 KB

    const int tid    = threadIdx.x;
    const int wave   = tid >> 5;
    const int lane   = tid & 31;
    const int laneLo = lane & 15;
    const int laneHi = lane >> 4;
    const int m0     = blockIdx.x * ROWS;
    const int vstart = blockIdx.y * VCHUNK;
    const int vend   = (vstart + VCHUNK) < VPAD ? (vstart + VCHUNK) : VPAD;

    if (tid < ROWS) tb[tid] = tbin[m0 + tid];
    for (int i = tid; i < ROWS * NHIST; i += 256)
        ((int*)seedL)[i] = seeds[(size_t)m0 * NHIST + i];
    __syncthreads();

    Frag a[NSETS][4];
#pragma unroll
    for (int s = 0; s < NSETS; ++s)
        load_frags(a[s], gN + (size_t)(m0 + s * 16 + laneLo) * DIM, laneHi);

    for (int vbase = vstart; vbase < vend; vbase += 128) {
        const int n = vbase + wave * 16 + laneLo;
        Frag b[4];
        load_frags(b, cN + (size_t)n * DIM, laneHi);
#pragma unroll
        for (int s = 0; s < NSETS; ++s) {
            v8f acc = {};
#pragma unroll
            for (int kk = 0; kk < 4; ++kk)
                acc = __builtin_amdgcn_wmma_f32_16x16x32_f16(
                        false, a[s][kk].f, false, b[kk].f, (short)0, acc, false, false);
            if (n < VOCAB) {
#pragma unroll
                for (int r = 0; r < 8; ++r) {
                    const int m = s * 16 + r + 8 * laneHi;
                    const float sc = acc[r];
                    if (score_bin(sc) >= tb[m]) {   // identical binning as pass 1
                        bool isSeed = false;
                        for (int h = 0; h < NHIST; ++h) isSeed |= (seedL[m][h] == n);
                        if (!isSeed) {
                            const int pos = atomicAdd(&cnt[m0 + m], 1);
                            if (pos < CAND_CAP) {
                                const size_t off = (size_t)(m0 + m) * CAND_CAP + pos;
                                candIdx[off]   = n;
                                candScore[off] = sc;
                            }
                        }
                    }
                }
            }
        }
    }
}

// ---------------------------------------------------------------------------
// Per-row bitonic sort of candidates, emit top-500 (indices then scores).
// One block per batch row.
// ---------------------------------------------------------------------------
__global__ __launch_bounds__(256) void sort_emit(const int* __restrict__ cnt,
                                                 const int* __restrict__ candIdx,
                                                 const float* __restrict__ candScore,
                                                 float* __restrict__ out) {
    __shared__ unsigned long long sortbuf[SORT_N];  // 8 KB
    const int tid = threadIdx.x;
    const int row = blockIdx.x;
    int C = cnt[row];
    C = C < CAND_CAP ? C : CAND_CAP;

    for (int i = tid; i < SORT_N; i += 256) {
        unsigned long long key = 0ull;  // pad: minimal key
        if (i < C) {
            const size_t off = (size_t)row * CAND_CAP + i;
            const unsigned int b  = __float_as_uint(candScore[off]);
            const unsigned int su = (b & 0x80000000u) ? ~b : (b | 0x80000000u);
            key = ((unsigned long long)su << 32) |
                  (unsigned long long)(~(unsigned int)candIdx[off]);
        }
        sortbuf[i] = key;
    }
    __syncthreads();
    // Bitonic sort, descending (score desc; equal score -> index asc via ~idx).
    for (int k = 2; k <= SORT_N; k <<= 1) {
        for (int j = k >> 1; j > 0; j >>= 1) {
            for (int t = tid; t < SORT_N / 2; t += 256) {
                const int i = ((t / j) * 2 * j) + (t % j);
                const int p = i + j;
                const bool asc = ((i & k) != 0);
                const unsigned long long x = sortbuf[i];
                const unsigned long long y = sortbuf[p];
                if (asc ? (x > y) : (x < y)) { sortbuf[i] = y; sortbuf[p] = x; }
            }
            __syncthreads();
        }
    }
    for (int i = tid; i < TOPK; i += 256) {
        const unsigned long long key = sortbuf[i];
        const unsigned int su  = (unsigned int)(key >> 32);
        const unsigned int idx = ~((unsigned int)(key & 0xFFFFFFFFull));
        const unsigned int b   = (su & 0x80000000u) ? (su ^ 0x80000000u) : ~su;
        out[(size_t)row * TOPK + i] = (float)idx;
        out[(size_t)BATCH * TOPK + (size_t)row * TOPK + i] = __uint_as_float(b);
    }
}

// ---------------------------------------------------------------------------
extern "C" void kernel_launch(void* const* d_in, const int* in_sizes, int n_in,
                              void* d_out, int out_size, void* d_ws, size_t ws_size,
                              hipStream_t stream) {
    (void)in_sizes; (void)n_in; (void)out_size; (void)ws_size;
    const float* gen   = (const float*)d_in[0];
    const float* cat   = (const float*)d_in[1];
    const int*   seeds = (const int*)d_in[2];

    // Workspace layout (~62.5 MB total):
    char* ws = (char*)d_ws;
    _Float16* gN = (_Float16*)ws;                                    // 1.0 MB
    _Float16* cN = (_Float16*)(ws + (size_t)BATCH * DIM * 2);        // 25.6 MB
    char* p = ws + (size_t)BATCH * DIM * 2 + (size_t)VPAD * DIM * 2;
    int*   candIdx   = (int*)p;                  p += (size_t)BATCH * CAND_CAP * 4;  // 16 MB
    float* candScore = (float*)p;                p += (size_t)BATCH * CAND_CAP * 4;  // 16 MB
    int*   ghist     = (int*)p;                  p += (size_t)BATCH * NBINS * 4;     // 2 MB
    int*   cnt       = (int*)p;                  p += (size_t)BATCH * 4;             // ghist+cnt zeroed together
    int*   tbin      = (int*)p;
    float* out = (float*)d_out;

    const int nzero = BATCH * NBINS + BATCH;  // ghist and cnt are contiguous

    normalize_f16<<<BATCH / 8, 256, 0, stream>>>(gen, gN, BATCH, BATCH);
    normalize_f16<<<VPAD / 8, 256, 0, stream>>>(cat, cN, VPAD, VOCAB);
    zero_ints<<<(nzero + 255) / 256, 256, 0, stream>>>(ghist, nzero);
    gemm_hist<<<dim3(BATCH / ROWS, VSPLIT), 256, 0, stream>>>(gN, cN, ghist);
    threshold_k<<<(BATCH + 255) / 256, 256, 0, stream>>>(gN, cN, seeds, ghist, tbin);
    collect<<<dim3(BATCH / ROWS, VSPLIT), 256, 0, stream>>>(gN, cN, seeds, tbin, cnt,
                                                            candIdx, candScore);
    sort_emit<<<BATCH, 256, 0, stream>>>(cnt, candIdx, candScore, out);
}
